// MOE_63522566308221
// MI455X (gfx1250) — matrix operand order repaired and checked by reading, not compiled, not run on hardware
//
#include <hip/hip_runtime.h>

// ---------------- problem constants ----------------
#define Bb    8
#define Tt    2048
#define Cc    768
#define Ee    8
#define Kk    2
#define Hh    3072
#define CAPE  640
#define MROWS (Bb*CAPE)        // 5120 rows per expert
#define BT    (Bb*Tt)          // 16384 tokens

// ---------------- WMMA types ----------------
typedef __attribute__((ext_vector_type(16))) __bf16 v16bf;
typedef __attribute__((ext_vector_type(8)))  float  v8f;

union FragBf { v16bf v; unsigned u[8]; };

__device__ __forceinline__ unsigned short f2bf(float f) {
    union { float f; unsigned u; } x; x.f = f;
    unsigned r = x.u + 0x7FFFu + ((x.u >> 16) & 1u);   // round-to-nearest-even
    return (unsigned short)(r >> 16);
}

// low 32 bits of a flat shared pointer == LDS byte offset (ISA: flat->LDS
// mapping truncates to addr[31:0])
__device__ __forceinline__ unsigned lds_off(const void* p) {
    return (unsigned)(unsigned long long)p;
}

// one per-lane 16B async copy global -> LDS (ASYNCcnt-tracked)
__device__ __forceinline__ void async_cp16(unsigned lds, unsigned long long gaddr) {
    asm volatile("global_load_async_to_lds_b128 %0, %1, off"
                 :: "v"(lds), "v"(gaddr) : "memory");
}

// ============================================================
// 1) Router: logits = x @ w_router + b ; softmax ; top-2
// ============================================================
__global__ __launch_bounds__(256) void moe_router(
    const float* __restrict__ x, const float* __restrict__ wr,
    const float* __restrict__ br,
    float* __restrict__ probs, int* __restrict__ eidx)
{
    const int lane = threadIdx.x & 31;
    const int wv   = threadIdx.x >> 5;
    const int row  = blockIdx.x * 8 + wv;            // grid = BT/8
    float acc[Ee];
#pragma unroll
    for (int e = 0; e < Ee; ++e) acc[e] = 0.f;
    for (int c = lane; c < Cc; c += 32) {
        float xv = x[(size_t)row * Cc + c];
        const float* w = wr + (size_t)c * Ee;
#pragma unroll
        for (int e = 0; e < Ee; ++e) acc[e] += xv * w[e];
    }
#pragma unroll
    for (int off = 16; off > 0; off >>= 1) {
#pragma unroll
        for (int e = 0; e < Ee; ++e) acc[e] += __shfl_down(acc[e], off, 32);
    }
    if (lane == 0) {
        float l[Ee];
        float mx = -1e30f;
#pragma unroll
        for (int e = 0; e < Ee; ++e) { l[e] = acc[e] + br[e]; mx = fmaxf(mx, l[e]); }
        float s = 0.f;
#pragma unroll
        for (int e = 0; e < Ee; ++e) { l[e] = __expf(l[e] - mx); s += l[e]; }
        float inv = 1.f / s;
#pragma unroll
        for (int e = 0; e < Ee; ++e) l[e] *= inv;
        int   e0 = 0; float p0 = l[0];
#pragma unroll
        for (int e = 1; e < Ee; ++e) if (l[e] > p0) { p0 = l[e]; e0 = e; }
        int   e1 = -1; float p1 = -1.f;
#pragma unroll
        for (int e = 0; e < Ee; ++e) if (e != e0 && l[e] > p1) { p1 = l[e]; e1 = e; }
        probs[row*2+0] = p0;  probs[row*2+1] = p1;
        eidx [row*2+0] = e0;  eidx [row*2+1] = e1;
    }
}

// ============================================================
// 2) k-major running capacity positions (matches jnp.cumsum order)
// ============================================================
__global__ void moe_pos(const int* __restrict__ eidx, int* __restrict__ pos)
{
    const int tid = threadIdx.x;          // <<<1, 64>>>
    const int b = tid >> 3, e = tid & 7;
    int cnt = 0;
    for (int k = 0; k < Kk; ++k)
        for (int t = 0; t < Tt; ++t) {
            int a = (b * Tt + t) * Kk + k;
            if (eidx[a] == e) { pos[a] = cnt; ++cnt; }
        }
}

// ============================================================
// 3) zero expert buffer + scatter tokens (bf16 convert)
// ============================================================
__global__ void moe_zero_u32(unsigned* __restrict__ p, size_t n)
{
    size_t i = (size_t)blockIdx.x * blockDim.x + threadIdx.x;
    size_t st = (size_t)gridDim.x * blockDim.x;
    for (; i < n; i += st) p[i] = 0u;
}

__global__ __launch_bounds__(256) void moe_scatter(
    const float* __restrict__ x, const int* __restrict__ eidx,
    const int* __restrict__ pos, unsigned short* __restrict__ xbuf)
{
    const int idx = blockIdx.x;                      // 0..BT*Kk-1
    const int b = idx / (Tt * Kk);
    const int rem = idx % (Tt * Kk);
    const int t = rem >> 1, k = rem & 1;
    const int a = (b * Tt + t) * Kk + k;
    const int e = eidx[a], p = pos[a];
    if (p >= CAPE) return;                           // dropped by capacity
    const size_t dst = ((size_t)e * MROWS + (size_t)b * CAPE + p) * Cc;
    const size_t src = ((size_t)b * Tt + t) * Cc;
    for (int c = threadIdx.x; c < Cc; c += 256) xbuf[dst + c] = f2bf(x[src + c]);
}

// ============================================================
// 3b) weight pre-pass: f32 [E][K][N] -> bf16 transposed [E][N][K]
//     makes B-tile staging a contiguous copy and the whole
//     113MB bf16 weight set L2-resident during the GEMM phase
// ============================================================
__global__ __launch_bounds__(256) void moe_w2bfT(
    const float* __restrict__ w, unsigned short* __restrict__ wT,
    int Kd, int Nd)
{
    const int e = blockIdx.y;
    const float* in = w + (size_t)e * Kd * Nd;
    unsigned short* out = wT + (size_t)e * Kd * Nd;
    const size_t total = (size_t)Kd * Nd;
    for (size_t o = (size_t)blockIdx.x * blockDim.x + threadIdx.x; o < total;
         o += (size_t)gridDim.x * blockDim.x) {
        size_t n = o / Kd, k = o % Kd;
        out[o] = f2bf(in[k * Nd + n]);
    }
}

// ============================================================
// WMMA fragment loaders (bf16 pairs packed in u32 LDS words)
//   A 16x32: VGPR v -> kpair = (v<4?0:8) + (v&3) + 4*(lane>=16)
//   B 32x16: VGPR v -> kpair = (v<4?0:8) + (v&3) + 4*(lane>=16)
// LDS pitch 20 words: rows stay 16B aligned for async B128 writes,
// and 20*m mod 64 is distinct for m=0..15 -> conflict-free frag reads
// ============================================================
#define LDSW 20

__device__ __forceinline__ v16bf loadA(const unsigned (*sA)[LDSW], int mbase,
                                       int mrow, int lhalf)
{
    FragBf f;
#pragma unroll
    for (int v = 0; v < 8; ++v) {
        int kp = ((v < 4) ? 0 : 8) + (v & 3) + 4 * lhalf;
        f.u[v] = sA[mbase + mrow][kp];
    }
    return f.v;
}
__device__ __forceinline__ v16bf loadB(const unsigned (*sB)[LDSW], int nbase,
                                       int mrow, int lhalf)
{
    FragBf f;
#pragma unroll
    for (int v = 0; v < 8; ++v) f.u[v] = sB[nbase + mrow][((v < 4) ? 0 : 8) + (v & 3) + 4 * lhalf];
    return f.v;
}

// issue one 128x32 bf16 tile (rows x 64B) as async B128 copies:
// 512 chunks over 256 threads -> 2 per thread (rows r and r+64)
__device__ __forceinline__ void issue_tile(
    unsigned ldsBase, const char* gRowBase, int rowPitchB,
    int srow, int sq, int k0)
{
    unsigned l0 = ldsBase + (unsigned)srow * (LDSW * 4) + sq * 16;
    unsigned l1 = l0 + 64u * (LDSW * 4);
    unsigned long long g0 = (unsigned long long)(gRowBase
                          + (size_t)srow * rowPitchB + k0 * 2 + sq * 16);
    unsigned long long g1 = g0 + (size_t)64 * rowPitchB;
    async_cp16(l0, g0);
    async_cp16(l1, g1);
}

// ============================================================
// 4) Fused fc+gate GEMM + SwiGLU, async double-buffered
//    act = bf16(silu((A Wf + bf)*(A Wg + bg)))
// ============================================================
__global__ __launch_bounds__(256) void moe_gemm_fcgate(
    const unsigned short* __restrict__ xbuf,
    const unsigned short* __restrict__ wfT, const unsigned short* __restrict__ wgT,
    const float* __restrict__ bfc, const float* __restrict__ bgt,
    unsigned short* __restrict__ act)
{
    const int e = blockIdx.y;
    const int NT = Hh / 128;                              // 24
    const int m0 = (blockIdx.x / NT) * 128;
    const int n0 = (blockIdx.x % NT) * 128;
    const char* Ab  = (const char*)(xbuf + (size_t)e * MROWS * Cc)
                    + (size_t)m0 * (Cc * 2);
    const char* Bfb = (const char*)(wfT + (size_t)e * Hh * Cc)
                    + (size_t)n0 * (Cc * 2);
    const char* Bgb = (const char*)(wgT + (size_t)e * Hh * Cc)
                    + (size_t)n0 * (Cc * 2);

    __shared__ unsigned sA [2][128][LDSW];
    __shared__ unsigned sBf[2][128][LDSW];
    __shared__ unsigned sBg[2][128][LDSW];

    const int tid = threadIdx.x, lane = tid & 31, wv = tid >> 5;
    const int wm = wv & 1, wn = wv >> 1;
    const int lhalf = lane >> 4, mrow = lane & 15;
    const int srow = tid >> 2, sq = tid & 3;

    v8f accF[4][2], accG[4][2];
#pragma unroll
    for (int i = 0; i < 4; ++i)
#pragma unroll
        for (int j = 0; j < 2; ++j) {
            v8f z = {0.f,0.f,0.f,0.f,0.f,0.f,0.f,0.f};
            accF[i][j] = z; accG[i][j] = z;
        }

    // prologue: stage k-tile 0 into buffer 0
    issue_tile(lds_off(&sA [0][0][0]), Ab,  Cc * 2, srow, sq, 0);
    issue_tile(lds_off(&sBf[0][0][0]), Bfb, Cc * 2, srow, sq, 0);
    issue_tile(lds_off(&sBg[0][0][0]), Bgb, Cc * 2, srow, sq, 0);

    const int nk = Cc / 32;                               // 24
    for (int it = 0; it < nk; ++it) {
        const int cur = it & 1;
        if (it + 1 < nk) {
            const int nxt = cur ^ 1, k1 = (it + 1) * 32;
            issue_tile(lds_off(&sA [nxt][0][0]), Ab,  Cc * 2, srow, sq, k1);
            issue_tile(lds_off(&sBf[nxt][0][0]), Bfb, Cc * 2, srow, sq, k1);
            issue_tile(lds_off(&sBg[nxt][0][0]), Bgb, Cc * 2, srow, sq, k1);
            // 6 next-stage copies in flight; in-order => stage `it` landed
            asm volatile("s_wait_asynccnt 0x6" ::: "memory");
        } else {
            asm volatile("s_wait_asynccnt 0x0" ::: "memory");
        }
        __syncthreads();

        v16bf aF[4], bF[2], bG[2];
#pragma unroll
        for (int i = 0; i < 4; ++i)
            aF[i] = loadA(sA[cur], wm * 64 + i * 16, mrow, lhalf);
#pragma unroll
        for (int j = 0; j < 2; ++j) {
            bF[j] = loadB(sBf[cur], wn * 32 + j * 16, mrow, lhalf);
            bG[j] = loadB(sBg[cur], wn * 32 + j * 16, mrow, lhalf);
        }
#pragma unroll
        for (int i = 0; i < 4; ++i)
#pragma unroll
            for (int j = 0; j < 2; ++j) {
                accF[i][j] = __builtin_amdgcn_wmma_f32_16x16x32_bf16(
                    false, aF[i], false, bF[j], (short)0, accF[i][j], false, false);
                accG[i][j] = __builtin_amdgcn_wmma_f32_16x16x32_bf16(
                    false, aF[i], false, bG[j], (short)0, accG[i][j], false, false);
            }
        __syncthreads();   // all waves done reading `cur` before it is re-filled
    }

    // SwiGLU epilogue, write bf16 activations
#pragma unroll
    for (int i = 0; i < 4; ++i)
#pragma unroll
        for (int j = 0; j < 2; ++j) {
            int n = n0 + wn * 32 + j * 16 + mrow;
            float bf = bfc[e * Hh + n], bg = bgt[e * Hh + n];
#pragma unroll
            for (int r = 0; r < 8; ++r) {
                int m = m0 + wm * 64 + i * 16 + r + 8 * lhalf;
                float h = accF[i][j][r] + bf;
                float g = accG[i][j][r] + bg;
                float sgl = h * g;
                float a = sgl / (1.f + __expf(-sgl));     // silu
                act[((size_t)e * MROWS + m) * Hh + n] = f2bf(a);
            }
        }
}

// ============================================================
// 5) Projection GEMM: eo = act @ Wp + bp   (f32 out), async 2-buffer
// ============================================================
__global__ __launch_bounds__(256) void moe_gemm_proj(
    const unsigned short* __restrict__ act,
    const unsigned short* __restrict__ wpT, const float* __restrict__ bpr,
    float* __restrict__ eo)
{
    const int e = blockIdx.y;
    const int NT = Cc / 128;                              // 6
    const int m0 = (blockIdx.x / NT) * 128;
    const int n0 = (blockIdx.x % NT) * 128;
    const char* Ab = (const char*)(act + (size_t)e * MROWS * Hh)
                   + (size_t)m0 * (Hh * 2);
    const char* Bb2 = (const char*)(wpT + (size_t)e * Cc * Hh)
                   + (size_t)n0 * (Hh * 2);

    __shared__ unsigned sA[2][128][LDSW];
    __shared__ unsigned sB[2][128][LDSW];

    const int tid = threadIdx.x, lane = tid & 31, wv = tid >> 5;
    const int wm = wv & 1, wn = wv >> 1;
    const int lhalf = lane >> 4, mrow = lane & 15;
    const int srow = tid >> 2, sq = tid & 3;

    v8f acc[4][2];
#pragma unroll
    for (int i = 0; i < 4; ++i)
#pragma unroll
        for (int j = 0; j < 2; ++j) {
            v8f z = {0.f,0.f,0.f,0.f,0.f,0.f,0.f,0.f};
            acc[i][j] = z;
        }

    issue_tile(lds_off(&sA[0][0][0]), Ab,  Hh * 2, srow, sq, 0);
    issue_tile(lds_off(&sB[0][0][0]), Bb2, Hh * 2, srow, sq, 0);

    const int nk = Hh / 32;                               // 96
    for (int it = 0; it < nk; ++it) {
        const int cur = it & 1;
        if (it + 1 < nk) {
            const int nxt = cur ^ 1, k1 = (it + 1) * 32;
            issue_tile(lds_off(&sA[nxt][0][0]), Ab,  Hh * 2, srow, sq, k1);
            issue_tile(lds_off(&sB[nxt][0][0]), Bb2, Hh * 2, srow, sq, k1);
            asm volatile("s_wait_asynccnt 0x4" ::: "memory");
        } else {
            asm volatile("s_wait_asynccnt 0x0" ::: "memory");
        }
        __syncthreads();

        v16bf aF[4], bF[2];
#pragma unroll
        for (int i = 0; i < 4; ++i)
            aF[i] = loadA(sA[cur], wm * 64 + i * 16, mrow, lhalf);
#pragma unroll
        for (int j = 0; j < 2; ++j)
            bF[j] = loadB(sB[cur], wn * 32 + j * 16, mrow, lhalf);
#pragma unroll
        for (int i = 0; i < 4; ++i)
#pragma unroll
            for (int j = 0; j < 2; ++j)
                acc[i][j] = __builtin_amdgcn_wmma_f32_16x16x32_bf16(
                    false, aF[i], false, bF[j], (short)0, acc[i][j], false, false);
        __syncthreads();
    }

#pragma unroll
    for (int i = 0; i < 4; ++i)
#pragma unroll
        for (int j = 0; j < 2; ++j) {
            int n = n0 + wn * 32 + j * 16 + mrow;
            float bp = bpr[e * Cc + n];
#pragma unroll
            for (int r = 0; r < 8; ++r) {
                int m = m0 + wm * 64 + i * 16 + r + 8 * lhalf;
                eo[((size_t)e * MROWS + m) * Cc + n] = acc[i][j][r] + bp;
            }
        }
}

// ============================================================
// 6) Combine: out = p0*eo[e0, b*CAP+min(pos0,CAP-1)] + p1*...
// ============================================================
__global__ __launch_bounds__(256) void moe_combine(
    const float* __restrict__ eo, const float* __restrict__ probs,
    const int* __restrict__ eidx, const int* __restrict__ pos,
    float* __restrict__ out)
{
    const int idx = blockIdx.x;                  // 0..BT-1
    const int b = idx / Tt;
    const int a = idx * 2;
    int e0 = eidx[a], e1 = eidx[a + 1];
    int p0 = pos[a],  p1 = pos[a + 1];
    p0 = p0 < CAPE ? p0 : CAPE - 1;
    p1 = p1 < CAPE ? p1 : CAPE - 1;
    const float w0 = probs[a], w1 = probs[a + 1];
    const size_t r0 = ((size_t)e0 * MROWS + (size_t)b * CAPE + p0) * Cc;
    const size_t r1 = ((size_t)e1 * MROWS + (size_t)b * CAPE + p1) * Cc;
    const size_t o  = (size_t)idx * Cc;
    for (int c = threadIdx.x; c < Cc; c += 256)
        out[o + c] = w0 * eo[r0 + c] + w1 * eo[r1 + c];
}

// ============================================================
// launch
// ============================================================
extern "C" void kernel_launch(void* const* d_in, const int* in_sizes, int n_in,
                              void* d_out, int out_size, void* d_ws, size_t ws_size,
                              hipStream_t stream)
{
    const float* x   = (const float*)d_in[0];
    const float* wr  = (const float*)d_in[1];
    const float* br  = (const float*)d_in[2];
    const float* wfc = (const float*)d_in[3];
    const float* bfc = (const float*)d_in[4];
    const float* wgt = (const float*)d_in[5];
    const float* bgt = (const float*)d_in[6];
    const float* wpr = (const float*)d_in[7];
    const float* bpr = (const float*)d_in[8];
    float* out = (float*)d_out;

    char* ws = (char*)d_ws;
    size_t off = 0;
    auto carve = [&](size_t bytes) -> void* {
        void* p = ws + off;
        off += (bytes + 255) & ~(size_t)255;
        return p;
    };
    float*          probs = (float*)carve((size_t)BT * Kk * sizeof(float));
    int*            eidx  = (int*)  carve((size_t)BT * Kk * sizeof(int));
    int*            pos   = (int*)  carve((size_t)BT * Kk * sizeof(int));
    unsigned short* xbuf  = (unsigned short*)carve((size_t)Ee * MROWS * Cc * sizeof(unsigned short));
    unsigned short* act   = (unsigned short*)carve((size_t)Ee * MROWS * Hh * sizeof(unsigned short));
    float*          eo    = (float*)carve((size_t)Ee * MROWS * Cc * sizeof(float));
    unsigned short* wfT   = (unsigned short*)carve((size_t)Ee * Cc * Hh * sizeof(unsigned short));
    unsigned short* wgT   = (unsigned short*)carve((size_t)Ee * Cc * Hh * sizeof(unsigned short));
    unsigned short* wpT   = (unsigned short*)carve((size_t)Ee * Hh * Cc * sizeof(unsigned short));

    moe_router <<<BT / 8, 256, 0, stream>>>(x, wr, br, probs, eidx);
    moe_pos    <<<1, 64, 0, stream>>>(eidx, pos);
    moe_zero_u32<<<4096, 256, 0, stream>>>((unsigned*)xbuf,
                                           (size_t)Ee * MROWS * Cc / 2);
    moe_scatter<<<BT * Kk, 256, 0, stream>>>(x, eidx, pos, xbuf);

    dim3 gw(2048, Ee);
    moe_w2bfT<<<gw, 256, 0, stream>>>(wfc, wfT, Cc, Hh);  // [C][H] -> [H][C]
    moe_w2bfT<<<gw, 256, 0, stream>>>(wgt, wgT, Cc, Hh);
    moe_w2bfT<<<gw, 256, 0, stream>>>(wpr, wpT, Hh, Cc);  // [H][C] -> [C][H]

    dim3 g1((MROWS / 128) * (Hh / 128), Ee);              // 40*24, 8
    moe_gemm_fcgate<<<g1, 256, 0, stream>>>(xbuf, wfT, wgT, bfc, bgt, act);

    dim3 g2((MROWS / 128) * (Cc / 128), Ee);              // 40*6, 8
    moe_gemm_proj<<<g2, 256, 0, stream>>>(act, wpT, bpr, eo);

    moe_combine<<<BT, 256, 0, stream>>>(eo, probs, eidx, pos, out);
}
